// Attention_41704132444356
// MI455X (gfx1250) — matrix operand order repaired and checked by reading, not compiled, not run on hardware
//
#include <hip/hip_runtime.h>

typedef __attribute__((ext_vector_type(16))) _Float16 v16h;
typedef __attribute__((ext_vector_type(8)))  _Float16 v8h;
typedef __attribute__((ext_vector_type(8)))  float    v8f;

constexpr int   B      = 2;
constexpr int   S      = 2048;
constexpr int   HQ     = 32;
constexpr int   HK     = 8;
constexpr int   D      = 128;
constexpr int   G      = HQ / HK;       // 4
constexpr int   WINDOW = 1024;
constexpr float THETA  = 10000.0f;
constexpr int   QTILE  = 16;            // query rows per wave (one WMMA M)
constexpr int   QBLK   = 32;            // query rows per workgroup (2 tiles)
constexpr int   KCHUNK = 32;            // keys per iteration (one WMMA K)
constexpr int   NWAVES = 8;             // 4 heads x 2 query tiles

__device__ __forceinline__ v8f wmma_f16(v16h a, v16h b, v8f c) {
  // (neg_a, A, neg_b, B, c_mod, C, reuse_a, reuse_b)
  return __builtin_amdgcn_wmma_f32_16x16x32_f16(false, a, false, b, (short)0, c,
                                                false, false);
}

__global__ __launch_bounds__(256) void fa_swa_gqa_kernel(
    const float* __restrict__ Q, const float* __restrict__ K,
    const float* __restrict__ V, const int* __restrict__ POS,
    float* __restrict__ OUT) {
  // K tile: [key][d] f16 (d-contiguous -> B-frags for Q*K^T)
  __shared__ __align__(32) _Float16 Kl[KCHUNK * D];
  // V tile transposed: [d][key] f16 (key-contiguous -> B-frags for P*V)
  __shared__ __align__(32) _Float16 Vt[D * KCHUNK];
  // Per-wave P scratch for D-frag -> A-frag layout conversion
  __shared__ __align__(32) _Float16 Pl[NWAVES][QTILE * KCHUNK];

  const int tid  = threadIdx.x;
  const int lane = tid & 31;
  const int wave = tid >> 5;
  const int ln   = lane & 15;  // N-col / A-row within half
  const int hi   = lane >> 4;  // half-wave select
  const int g    = wave & 3;   // query head within group
  const int qt   = wave >> 2;  // which 16-row query tile

  const int nqb = S / QBLK;
  const int qb  = blockIdx.x % nqb;
  const int hk  = (blockIdx.x / nqb) % HK;
  const int b   = blockIdx.x / (nqb * HK);
  const int h   = hk * G + g;
  const int q0  = qb * QBLK;        // workgroup's first query row
  const int q0w = q0 + qt * QTILE;  // this wave's first query row

  // ---------------- Q load + RoPE + scale -> A fragments ----------------
  // A-frag (16x32 f16): lane row M = ln; per slice s (d = s*32..s*32+31):
  //   halves 0..7  -> d = s*32 + hi*8 + (0..7)
  //   halves 8..15 -> d = s*32 + 16 + hi*8 + (0..7)
  const int    qtok = b * S + q0w + ln;
  const float  posq = (float)POS[qtok];
  const float* qp   = Q + (size_t)qtok * (HQ * D) + h * D;

  float qbuf[4][16];
#pragma unroll
  for (int s = 0; s < 4; ++s) {
    const int d0 = s * 32 + hi * 8;
#pragma unroll
    for (int t = 0; t < 8; ++t) {
      qbuf[s][t]     = qp[d0 + t];
      qbuf[s][t + 8] = qp[d0 + 16 + t];
    }
  }
  // RoPE: pair (d, d+64) lives at (slice s, slot t) and (slice s+2, slot t)
#pragma unroll
  for (int s = 0; s < 2; ++s) {
#pragma unroll
    for (int t = 0; t < 16; ++t) {
      const int   dd  = s * 32 + hi * 8 + ((t < 8) ? t : (t + 8));
      const float ang = posq * __powf(THETA, -(float)dd * (1.0f / 64.0f));
      float c, sn;
      __sincosf(ang, &c, &sn);
      const float x1 = qbuf[s][t], x2 = qbuf[s + 2][t];
      qbuf[s][t]     = x1 * c - x2 * sn;
      qbuf[s + 2][t] = x2 * c + x1 * sn;
    }
  }
  const float scale = 0.08838834764831845f;  // 1/sqrt(128)
  v16h qA[4];
#pragma unroll
  for (int s = 0; s < 4; ++s)
#pragma unroll
    for (int t = 0; t < 16; ++t) qA[s][t] = (_Float16)(qbuf[s][t] * scale);

  // all-ones B fragment: P x ones broadcasts row-sums into a D tile
  v16h ones;
#pragma unroll
  for (int t = 0; t < 16; ++t) ones[t] = (_Float16)1.0f;

  // ---------------- accumulator / online-softmax state ----------------
  // D-frag rows per lane: m = r + 8*hi, col n = ln
  v8f acc[8];
#pragma unroll
  for (int db = 0; db < 8; ++db)
#pragma unroll
    for (int r = 0; r < 8; ++r) acc[db][r] = 0.0f;
  float mrow[8], lrow[8];
#pragma unroll
  for (int r = 0; r < 8; ++r) { mrow[r] = -1e30f; lrow[r] = 0.0f; }

  // K-load: with 256 threads, dp = tid & 63 is loop-invariant -> hoist freq
  const int   kdp   = tid & 63;
  const float kinvf = __powf(THETA, -(float)kdp * (1.0f / 64.0f));
  const int   vd    = tid & 127;  // V-load d index, also invariant

  int klo0 = q0 - (WINDOW - 1);
  if (klo0 < 0) klo0 = 0;
  const int klo = klo0 & ~(KCHUNK - 1);
  const int khi = q0 + QBLK;

  for (int kc = klo; kc < khi; kc += KCHUNK) {
    __syncthreads();  // protect LDS tiles from previous iteration's readers

    // ---- cooperative K tile load + RoPE (f32 -> f16, key-major) ----
#pragma unroll
    for (int j = 0; j < 8; ++j) {
      const int    kl  = (tid >> 6) + j * 4;  // key 0..31
      const int    tok = b * S + kc + kl;
      const float* kp  = K + (size_t)tok * (HK * D) + hk * D;
      const float  x1 = kp[kdp], x2 = kp[kdp + 64];
      const float  ang = (float)POS[tok] * kinvf;
      float c, sn;
      __sincosf(ang, &c, &sn);
      Kl[kl * D + kdp]      = (_Float16)(x1 * c - x2 * sn);
      Kl[kl * D + kdp + 64] = (_Float16)(x2 * c + x1 * sn);
    }
    // ---- cooperative V tile load, transposed into [d][key] ----
#pragma unroll
    for (int j = 0; j < 16; ++j) {
      const int kl  = (tid >> 7) + j * 2;  // key 0..31
      const int tok = b * S + kc + kl;
      Vt[vd * KCHUNK + kl] =
          (_Float16)(V[(size_t)tok * (HK * D) + hk * D + vd]);
    }
    __syncthreads();

    // skip compute when this wave's rows can't see any key of this chunk
    // (wave-uniform branch; EXEC stays full for the WMMA region)
    if (kc > q0w + (QTILE - 1) || kc + KCHUNK <= q0w - (WINDOW - 1)) continue;

    // ---- scores: S = Q * K^T, two 16x16 tiles, 4 d-slices each ----
    v8f st[2];
#pragma unroll
    for (int nt = 0; nt < 2; ++nt) {
      v8f sc;
#pragma unroll
      for (int r = 0; r < 8; ++r) sc[r] = 0.0f;
#pragma unroll
      for (int s = 0; s < 4; ++s) {
        // B-frag: lane n = ln (key), k = hi*8 + (0..7), 16 + hi*8 + (0..7)
        const v16h bf = __builtin_shufflevector(
            *(const v8h*)&Kl[(nt * 16 + ln) * D + s * 32 + hi * 8],
            *(const v8h*)&Kl[(nt * 16 + ln) * D + s * 32 + 16 + hi * 8],
            0, 1, 2, 3, 4, 5, 6, 7, 8, 9, 10, 11, 12, 13, 14, 15);
        sc = wmma_f16(qA[s], bf, sc);
      }
      st[nt] = sc;
    }

    // ---- online softmax (rows r + 8*hi, cols = key lanes) ----
    const int j0 = kc + ln, j1 = j0 + 16;
    float alpha[8];
#pragma unroll
    for (int r = 0; r < 8; ++r) {
      const int irow = q0w + 8 * hi + r;
      // causal + sliding window in one unsigned compare: 0 <= i-j < WINDOW
      const bool ok0 = (unsigned)(irow - j0) < (unsigned)WINDOW;
      const bool ok1 = (unsigned)(irow - j1) < (unsigned)WINDOW;
      const float e0 = ok0 ? st[0][r] : -1e30f;
      const float e1 = ok1 ? st[1][r] : -1e30f;
      float mx = fmaxf(e0, e1);
#pragma unroll
      for (int m = 1; m < 16; m <<= 1) mx = fmaxf(mx, __shfl_xor(mx, m, 32));
      mx = fmaxf(mx, mrow[r]);
      alpha[r] = __expf(mrow[r] - mx);
      mrow[r]  = mx;
      const float p0 = ok0 ? __expf(e0 - mx) : 0.0f;
      const float p1 = ok1 ? __expf(e1 - mx) : 0.0f;
#pragma unroll
      for (int db = 0; db < 8; ++db) acc[db][r] *= alpha[r];
      // stage P (f16) for D-frag -> A-frag conversion
      Pl[wave][(8 * hi + r) * KCHUNK + ln]      = (_Float16)p0;
      Pl[wave][(8 * hi + r) * KCHUNK + 16 + ln] = (_Float16)p1;
    }

    // ---- P A-frag (same wave produced it; compiler orders via DScnt) ----
    union { v16h f; v8h h2[2]; } pa;
    pa.h2[0] = *(const v8h*)&Pl[wave][ln * KCHUNK + hi * 8];
    pa.h2[1] = *(const v8h*)&Pl[wave][ln * KCHUNK + 16 + hi * 8];

    // ---- row sums via WMMA: (P x ones) broadcasts rowsum to all lanes ----
    v8f zero;
#pragma unroll
    for (int r = 0; r < 8; ++r) zero[r] = 0.0f;
    const v8f srow = wmma_f16(pa.f, ones, zero);
#pragma unroll
    for (int r = 0; r < 8; ++r) lrow[r] = lrow[r] * alpha[r] + srow[r];

    // ---- out += P * V over 8 d-column blocks ----
#pragma unroll
    for (int db = 0; db < 8; ++db) {
      // B-frag: lane n = ln (d-col), k = hi*8 + (0..7), 16 + hi*8 + (0..7)
      const v16h vf = __builtin_shufflevector(
          *(const v8h*)&Vt[(db * 16 + ln) * KCHUNK + hi * 8],
          *(const v8h*)&Vt[(db * 16 + ln) * KCHUNK + 16 + hi * 8],
          0, 1, 2, 3, 4, 5, 6, 7, 8, 9, 10, 11, 12, 13, 14, 15);
      acc[db] = wmma_f16(pa.f, vf, acc[db]);
    }
  }

  // ---------------- normalize and store (f32 output) ----------------
#pragma unroll
  for (int r = 0; r < 8; ++r) {
    const float inv = 1.0f / lrow[r];
    float* op = OUT + (size_t)(b * S + q0w + 8 * hi + r) * (HQ * D) + h * D;
#pragma unroll
    for (int db = 0; db < 8; ++db) op[db * 16 + ln] = acc[db][r] * inv;
  }
}

extern "C" void kernel_launch(void* const* d_in, const int* in_sizes, int n_in,
                              void* d_out, int out_size, void* d_ws,
                              size_t ws_size, hipStream_t stream) {
  const float* q   = (const float*)d_in[0];
  const float* k   = (const float*)d_in[1];
  const float* v   = (const float*)d_in[2];
  const int*   pos = (const int*)d_in[3];
  float*       out = (float*)d_out;

  // one block per (batch, kv-head, 32-query block);
  // 8 waves = 4 query heads x 2 query tiles sharing the LDS K/V tiles
  dim3 grid(B * HK * (S / QBLK));
  dim3 block(256);
  fa_swa_gqa_kernel<<<grid, block, 0, stream>>>(q, k, v, pos, out);
}
